// PhyloEGNN_67954972557873
// MI455X (gfx1250) — matrix operand order repaired and checked by reading, not compiled, not run on hardware
//
#include <hip/hip_runtime.h>
#include <hip/hip_bf16.h>

typedef __attribute__((ext_vector_type(16))) _Float16 v16h;
typedef __attribute__((ext_vector_type(8)))  _Float16 v8h;
typedef __attribute__((ext_vector_type(8)))  float    v8f;

#define N_NODES  50000
#define N_EDGES  800000
#define N_GRAPHS 64

// fast silu/sigmoid: bare v_rcp_f32 (hardware intrinsic) instead of IEEE
// div_scale/fma fixup chain -- __fdividef does NOT lower to v_rcp on AMD.
__device__ __forceinline__ float rcp_fast(float x)  { return __builtin_amdgcn_rcpf(x); }
__device__ __forceinline__ float silu_f(float x)    { return x * rcp_fast(1.f + __expf(-x)); }
__device__ __forceinline__ float sigmoid_f(float x) { return rcp_fast(1.f + __expf(-x)); }

// ---------------------------------------------------------------------------
// A-fragment load: 16x32 f16 tile from row-major LDS tile (stride in halfs).
// ISA 7.12.2: lanes 0-15 = row M=lane, K=0..7 (v0-3) and 16..23 (v4-7);
//             lanes 16-31 = row M=lane-16, K=8..15 and 24..31.
__device__ __forceinline__ v16h load_a_frag(const _Float16* tile, int stride,
                                            int mrow, int hr, int k0) {
  const _Float16* base = tile + mrow * stride + k0 + 8 * hr;
  v8h lo = *(const v8h*)base;
  v8h hi = *(const v8h*)(base + 16);
  return __builtin_shufflevector(lo, hi, 0,1,2,3,4,5,6,7,8,9,10,11,12,13,14,15);
}

// ---------------------------------------------------------------------------
// Pack fp32 weight [Kreal x Nreal] (row-major, K-major like the reference
// x @ W) into f16 B-fragments (32x16 per fragment) for v_wmma_f32_16x16x32_f16.
// B layout: lane L holds column N=(L&15); half-group hr=L>>4 covers K 8*hr..+7 and 16+8*hr..+7,
// element i -> K = kc*32 + 8*hr + i (i<8), kc*32 + 16 + 8*hr + (i-8) (i>=8).  Fragment index = nt*kchunks + kc.
__global__ void pack_w_kernel(const float* __restrict__ W, _Float16* __restrict__ out,
                              int Kreal, int Nreal, int kchunks, int ntiles) {
  int t = blockIdx.x * blockDim.x + threadIdx.x;
  int nfrag = kchunks * ntiles;
  if (t >= nfrag * 32) return;
  int lane = t & 31;
  int frag = t >> 5;
  int kc = frag % kchunks;
  int nt = frag / kchunks;
  int hr = lane >> 4;
  int n  = nt * 16 + (lane & 15);
  _Float16* dst = out + (size_t)frag * 512 + lane * 16;
#pragma unroll
  for (int i = 0; i < 16; ++i) {
    int K = kc * 32 + (i < 8 ? 8 * hr + i : 16 + 8 * hr + (i - 8));
    float v = (K < Kreal && n < Nreal) ? W[(size_t)K * Nreal + n] : 0.f;
    dst[i] = (_Float16)v;
  }
}

// ---------------------------------------------------------------------------
__global__ void zero_kernel(float* __restrict__ p, int n) {
  int t = blockIdx.x * blockDim.x + threadIdx.x;
  if (t < n) p[t] = 0.f;
}
__global__ void copy_kernel(float* __restrict__ dst, const float* __restrict__ src, int n) {
  int t = blockIdx.x * blockDim.x + threadIdx.x;
  if (t < n) dst[t] = src[t];
}
__global__ void addinto_kernel(float* __restrict__ dst, const float* __restrict__ src, int n) {
  int t = blockIdx.x * blockDim.x + threadIdx.x;
  if (t < n) dst[t] += src[t];
}

// ---------------------------------------------------------------------------
// input projection: h = x @ W(32x64) + b
__global__ __launch_bounds__(256)
void input_proj_kernel(const float* __restrict__ x, const float* __restrict__ W,
                       const float* __restrict__ b, float* __restrict__ h) {
  __shared__ float Ws[32 * 64];
  __shared__ float bs[64];
  for (int i = threadIdx.x; i < 32 * 64; i += 256) Ws[i] = W[i];
  if (threadIdx.x < 64) bs[threadIdx.x] = b[threadIdx.x];
  __syncthreads();
  int t = blockIdx.x * blockDim.x + threadIdx.x;
  if (t >= N_NODES * 64) return;
  int n = t >> 6, j = t & 63;
  float acc = bs[j];
  const float* xr = x + (size_t)n * 32;
#pragma unroll
  for (int k = 0; k < 32; ++k) acc += xr[k] * Ws[k * 64 + j];
  h[t] = acc;
}

// ---------------------------------------------------------------------------
// Coord update: delta = coord2(silu(coord1([h_r|h_c|rel]))); normalize, gate,
// scatter-add into cdelta[row].  One wave = 16 edges.  131 padded to 160.
__global__ __launch_bounds__(256)
void coord_kernel(const float* __restrict__ h, const float* __restrict__ coords,
                  const int* __restrict__ row, const int* __restrict__ col,
                  const _Float16* __restrict__ w1p, const float* __restrict__ b1,
                  const _Float16* __restrict__ w2p, const float* __restrict__ b2,
                  const float* __restrict__ ewcW, const float* __restrict__ ewcB,
                  const float* __restrict__ cscale, float* __restrict__ cdelta) {
  __shared__ __align__(16) _Float16 tiles[8][16][160];
  __shared__ float relbuf[8][16][3];
  __shared__ float dbuf[8][16][4];
  int wave = threadIdx.x >> 5, lane = threadIdx.x & 31;
  int e0 = (blockIdx.x * 8 + wave) * 16;
  _Float16 (*tile)[160] = tiles[wave];

  __builtin_prefetch(w1p + lane * 16, 0, 3);

  for (int m = 0; m < 16; ++m) {
    int e = e0 + m;
    int r = row[e], c = col[e];
    tile[m][lane]        = (_Float16)h[(size_t)r * 64 + lane];
    tile[m][32 + lane]   = (_Float16)h[(size_t)r * 64 + 32 + lane];
    tile[m][64 + lane]   = (_Float16)h[(size_t)c * 64 + lane];
    tile[m][96 + lane]   = (_Float16)h[(size_t)c * 64 + 32 + lane];
    if (lane < 3) {
      float rl = coords[(size_t)r * 3 + lane] - coords[(size_t)c * 3 + lane];
      relbuf[wave][m][lane] = rl;
      tile[m][128 + lane] = (_Float16)rl;
    } else {
      tile[m][128 + lane] = (_Float16)0.f;   // pad cols 131..159
    }
  }
  int hr = lane >> 4, mrow = lane & 15;

  v16h A[5];
#pragma unroll
  for (int kc = 0; kc < 5; ++kc) A[kc] = load_a_frag(&tile[0][0], 160, mrow, hr, kc * 32);

  // GEMM1: 160 -> 128, silu, back to LDS tile (f16) for GEMM2.
  // Batch all K-chunk B-fragments per N-tile so loads clause & overlap the WMMAs.
#pragma unroll
  for (int nt = 0; nt < 8; ++nt) {
    v16h Bf[5];
#pragma unroll
    for (int kc = 0; kc < 5; ++kc)
      Bf[kc] = *(const v16h*)(w1p + (size_t)(nt * 5 + kc) * 512 + lane * 16);
    if (nt + 1 < 8)
      __builtin_prefetch(w1p + (size_t)((nt + 1) * 5) * 512 + lane * 16, 0, 3);
    float bias = b1[nt * 16 + mrow];
    v8f C;
#pragma unroll
    for (int v = 0; v < 8; ++v) C[v] = bias;
#pragma unroll
    for (int kc = 0; kc < 5; ++kc)
      C = __builtin_amdgcn_wmma_f32_16x16x32_f16(false, A[kc], false, Bf[kc], (short)0, C, false, false);
#pragma unroll
    for (int v = 0; v < 8; ++v)
      tile[v + 8 * hr][nt * 16 + mrow] = (_Float16)silu_f(C[v]);
  }

  v16h A2[4];
#pragma unroll
  for (int kc = 0; kc < 4; ++kc) A2[kc] = load_a_frag(&tile[0][0], 160, mrow, hr, kc * 32);

  // GEMM2: 128 -> 3 (padded to 16)
  {
    v16h Bf[4];
#pragma unroll
    for (int kc = 0; kc < 4; ++kc)
      Bf[kc] = *(const v16h*)(w2p + (size_t)kc * 512 + lane * 16);
    float bias2 = (mrow < 3) ? b2[mrow] : 0.f;
    v8f C2;
#pragma unroll
    for (int v = 0; v < 8; ++v) C2[v] = bias2;
#pragma unroll
    for (int kc = 0; kc < 4; ++kc)
      C2 = __builtin_amdgcn_wmma_f32_16x16x32_f16(false, A2[kc], false, Bf[kc], (short)0, C2, false, false);
#pragma unroll
    for (int v = 0; v < 8; ++v)
      if (mrow < 3) dbuf[wave][v + 8 * hr][mrow] = C2[v];
  }

  // per-edge epilogue: normalize, gate with sigmoid(ewc(rel)), scatter
  if (lane < 16) {
    int e = e0 + lane;
    float dx = dbuf[wave][lane][0], dy = dbuf[wave][lane][1], dz = dbuf[wave][lane][2];
    float nrm = fmaxf(sqrtf(dx * dx + dy * dy + dz * dz), 1e-8f);
    float rx = relbuf[wave][lane][0], ry = relbuf[wave][lane][1], rz = relbuf[wave][lane][2];
    float sig = sigmoid_f(rx * ewcW[0] + ry * ewcW[1] + rz * ewcW[2] + ewcB[0]);
    float sc = cscale[0] * sig * rcp_fast(nrm);
    int r = row[e];
    atomicAdd(&cdelta[(size_t)r * 3 + 0], dx * sc);
    atomicAdd(&cdelta[(size_t)r * 3 + 1], dy * sc);
    atomicAdd(&cdelta[(size_t)r * 3 + 2], dz * sc);
  }
}

// ---------------------------------------------------------------------------
// Edge messages: e_ij = silu(edge2(silu(edge1([h_r|h_c|rel])))); scatter into agg[row].
__global__ __launch_bounds__(256)
void edge_msg_kernel(const float* __restrict__ h, const float* __restrict__ coords,
                     const int* __restrict__ row, const int* __restrict__ col,
                     const _Float16* __restrict__ w1p, const float* __restrict__ b1,
                     const _Float16* __restrict__ w2p, const float* __restrict__ b2,
                     float* __restrict__ agg) {
  __shared__ __align__(16) _Float16 tiles[8][16][160];
  int wave = threadIdx.x >> 5, lane = threadIdx.x & 31;
  int e0 = (blockIdx.x * 8 + wave) * 16;
  _Float16 (*tile)[160] = tiles[wave];

  __builtin_prefetch(w1p + lane * 16, 0, 3);

  for (int m = 0; m < 16; ++m) {
    int e = e0 + m;
    int r = row[e], c = col[e];
    tile[m][lane]        = (_Float16)h[(size_t)r * 64 + lane];
    tile[m][32 + lane]   = (_Float16)h[(size_t)r * 64 + 32 + lane];
    tile[m][64 + lane]   = (_Float16)h[(size_t)c * 64 + lane];
    tile[m][96 + lane]   = (_Float16)h[(size_t)c * 64 + 32 + lane];
    if (lane < 3)
      tile[m][128 + lane] = (_Float16)(coords[(size_t)r * 3 + lane] - coords[(size_t)c * 3 + lane]);
    else
      tile[m][128 + lane] = (_Float16)0.f;
  }
  int hr = lane >> 4, mrow = lane & 15;

  v16h A[5];
#pragma unroll
  for (int kc = 0; kc < 5; ++kc) A[kc] = load_a_frag(&tile[0][0], 160, mrow, hr, kc * 32);

#pragma unroll
  for (int nt = 0; nt < 8; ++nt) {
    v16h Bf[5];
#pragma unroll
    for (int kc = 0; kc < 5; ++kc)
      Bf[kc] = *(const v16h*)(w1p + (size_t)(nt * 5 + kc) * 512 + lane * 16);
    if (nt + 1 < 8)
      __builtin_prefetch(w1p + (size_t)((nt + 1) * 5) * 512 + lane * 16, 0, 3);
    float bias = b1[nt * 16 + mrow];
    v8f C;
#pragma unroll
    for (int v = 0; v < 8; ++v) C[v] = bias;
#pragma unroll
    for (int kc = 0; kc < 5; ++kc)
      C = __builtin_amdgcn_wmma_f32_16x16x32_f16(false, A[kc], false, Bf[kc], (short)0, C, false, false);
#pragma unroll
    for (int v = 0; v < 8; ++v)
      tile[v + 8 * hr][nt * 16 + mrow] = (_Float16)silu_f(C[v]);
  }

  v16h A2[4];
#pragma unroll
  for (int kc = 0; kc < 4; ++kc) A2[kc] = load_a_frag(&tile[0][0], 160, mrow, hr, kc * 32);

  int rr[8];
#pragma unroll
  for (int v = 0; v < 8; ++v) rr[v] = row[e0 + v + 8 * hr];

#pragma unroll
  for (int nt = 0; nt < 4; ++nt) {
    v16h Bf[4];
#pragma unroll
    for (int kc = 0; kc < 4; ++kc)
      Bf[kc] = *(const v16h*)(w2p + (size_t)(nt * 4 + kc) * 512 + lane * 16);
    if (nt + 1 < 4)
      __builtin_prefetch(w2p + (size_t)((nt + 1) * 4) * 512 + lane * 16, 0, 3);
    float bias = b2[nt * 16 + mrow];
    v8f C;
#pragma unroll
    for (int v = 0; v < 8; ++v) C[v] = bias;
#pragma unroll
    for (int kc = 0; kc < 4; ++kc)
      C = __builtin_amdgcn_wmma_f32_16x16x32_f16(false, A2[kc], false, Bf[kc], (short)0, C, false, false);
#pragma unroll
    for (int v = 0; v < 8; ++v)
      atomicAdd(&agg[(size_t)rr[v] * 64 + nt * 16 + mrow], silu_f(C[v]));
  }
}

// ---------------------------------------------------------------------------
// Node update: h += node2(silu(node1([h|agg]))).  One wave = 16 nodes.
__global__ __launch_bounds__(256)
void node_kernel(float* __restrict__ h, const float* __restrict__ agg,
                 const _Float16* __restrict__ w1p, const float* __restrict__ b1,
                 const _Float16* __restrict__ w2p, const float* __restrict__ b2) {
  __shared__ __align__(16) _Float16 tiles[8][16][128];
  int wave = threadIdx.x >> 5, lane = threadIdx.x & 31;
  int n0 = (blockIdx.x * 8 + wave) * 16;
  if (n0 >= N_NODES) return;   // no cross-wave barriers used
  _Float16 (*tile)[128] = tiles[wave];

  __builtin_prefetch(w1p + lane * 16, 0, 3);

  for (int m = 0; m < 16; ++m) {
    size_t n = (size_t)(n0 + m);
    tile[m][lane]        = (_Float16)h[n * 64 + lane];
    tile[m][32 + lane]   = (_Float16)h[n * 64 + 32 + lane];
    tile[m][64 + lane]   = (_Float16)agg[n * 64 + lane];
    tile[m][96 + lane]   = (_Float16)agg[n * 64 + 32 + lane];
  }
  int hr = lane >> 4, mrow = lane & 15;

  v16h A[4];
#pragma unroll
  for (int kc = 0; kc < 4; ++kc) A[kc] = load_a_frag(&tile[0][0], 128, mrow, hr, kc * 32);

#pragma unroll
  for (int nt = 0; nt < 8; ++nt) {
    v16h Bf[4];
#pragma unroll
    for (int kc = 0; kc < 4; ++kc)
      Bf[kc] = *(const v16h*)(w1p + (size_t)(nt * 4 + kc) * 512 + lane * 16);
    if (nt + 1 < 8)
      __builtin_prefetch(w1p + (size_t)((nt + 1) * 4) * 512 + lane * 16, 0, 3);
    float bias = b1[nt * 16 + mrow];
    v8f C;
#pragma unroll
    for (int v = 0; v < 8; ++v) C[v] = bias;
#pragma unroll
    for (int kc = 0; kc < 4; ++kc)
      C = __builtin_amdgcn_wmma_f32_16x16x32_f16(false, A[kc], false, Bf[kc], (short)0, C, false, false);
#pragma unroll
    for (int v = 0; v < 8; ++v)
      tile[v + 8 * hr][nt * 16 + mrow] = (_Float16)silu_f(C[v]);
  }

  v16h A2[4];
#pragma unroll
  for (int kc = 0; kc < 4; ++kc) A2[kc] = load_a_frag(&tile[0][0], 128, mrow, hr, kc * 32);

#pragma unroll
  for (int nt = 0; nt < 4; ++nt) {
    v16h Bf[4];
#pragma unroll
    for (int kc = 0; kc < 4; ++kc)
      Bf[kc] = *(const v16h*)(w2p + (size_t)(nt * 4 + kc) * 512 + lane * 16);
    if (nt + 1 < 4)
      __builtin_prefetch(w2p + (size_t)((nt + 1) * 4) * 512 + lane * 16, 0, 3);
    float bias = b2[nt * 16 + mrow];
    v8f C;
#pragma unroll
    for (int v = 0; v < 8; ++v) C[v] = bias;
#pragma unroll
    for (int kc = 0; kc < 4; ++kc)
      C = __builtin_amdgcn_wmma_f32_16x16x32_f16(false, A2[kc], false, Bf[kc], (short)0, C, false, false);
#pragma unroll
    for (int v = 0; v < 8; ++v) {
      int m = v + 8 * hr;
      h[(size_t)(n0 + m) * 64 + nt * 16 + mrow] += C[v];   // residual, no activation
    }
  }
}

// ---------------------------------------------------------------------------
// Gate MLP: gate3(relu(gate2(relu(LN(gate1(h)))))) -> gate[n].  One wave = 1 node.
__global__ __launch_bounds__(256)
void gate_kernel(const float* __restrict__ h,
                 const float* __restrict__ g1W, const float* __restrict__ g1b,
                 const float* __restrict__ lng, const float* __restrict__ lnb,
                 const float* __restrict__ g2W, const float* __restrict__ g2b,
                 const float* __restrict__ g3W, const float* __restrict__ g3b,
                 float* __restrict__ gate) {
  __shared__ float W1[64 * 64];
  __shared__ float W2[64 * 32];
  __shared__ float W3[32];
  __shared__ float B1[64], B2[32], G[64], Bt[64];
  __shared__ float t1s[8][64];
  for (int i = threadIdx.x; i < 64 * 64; i += 256) W1[i] = g1W[i];
  for (int i = threadIdx.x; i < 64 * 32; i += 256) W2[i] = g2W[i];
  if (threadIdx.x < 32) { W3[threadIdx.x] = g3W[threadIdx.x]; B2[threadIdx.x] = g2b[threadIdx.x]; }
  if (threadIdx.x < 64) {
    B1[threadIdx.x] = g1b[threadIdx.x];
    G[threadIdx.x]  = lng[threadIdx.x];
    Bt[threadIdx.x] = lnb[threadIdx.x];
  }
  __syncthreads();
  int wave = threadIdx.x >> 5, lane = threadIdx.x & 31;
  int n = blockIdx.x * 8 + wave;
  if (n >= N_NODES) return;
  const float* hn = h + (size_t)n * 64;
  float a0 = B1[lane], a1 = B1[lane + 32];
  for (int k = 0; k < 64; ++k) {
    float hv = hn[k];
    a0 += hv * W1[k * 64 + lane];
    a1 += hv * W1[k * 64 + lane + 32];
  }
  float sum = a0 + a1, sq = a0 * a0 + a1 * a1;
#pragma unroll
  for (int o = 16; o >= 1; o >>= 1) {
    sum += __shfl_xor(sum, o, 32);
    sq  += __shfl_xor(sq, o, 32);
  }
  float mu = sum * (1.f / 64.f);
  float var = sq * (1.f / 64.f) - mu * mu;
  float inv = rsqrtf(var + 1e-5f);
  float v0 = fmaxf((a0 - mu) * inv * G[lane]      + Bt[lane],      0.f);
  float v1 = fmaxf((a1 - mu) * inv * G[lane + 32] + Bt[lane + 32], 0.f);
  t1s[wave][lane] = v0;
  t1s[wave][lane + 32] = v1;
  float t2 = B2[lane];
  for (int k = 0; k < 64; ++k) t2 += t1s[wave][k] * W2[k * 32 + lane];
  t2 = fmaxf(t2, 0.f);
  float gsum = t2 * W3[lane];
#pragma unroll
  for (int o = 16; o >= 1; o >>= 1) gsum += __shfl_xor(gsum, o, 32);
  if (lane == 0) gate[n] = gsum + g3b[0];
}

// ---------------------------------------------------------------------------
__global__ void initpool_kernel(int* __restrict__ mkey, float* __restrict__ s,
                                float* __restrict__ pooled) {
  int t = blockIdx.x * blockDim.x + threadIdx.x;
  if (t < N_GRAPHS) { mkey[t] = (int)0x80000000; s[t] = 0.f; }
  if (t < N_GRAPHS * 64) pooled[t] = 0.f;
}

__device__ __forceinline__ int f2key(float f) {
  int b = __float_as_int(f);
  return b >= 0 ? b : (b ^ 0x7fffffff);
}
__device__ __forceinline__ float key2f(int k) {
  return __int_as_float(k >= 0 ? k : (k ^ 0x7fffffff));
}

__global__ void segmax_kernel(const float* __restrict__ gate, const int* __restrict__ batch,
                              int* __restrict__ mkey) {
  int n = blockIdx.x * blockDim.x + threadIdx.x;
  if (n >= N_NODES) return;
  atomicMax(&mkey[batch[n]], f2key(gate[n]));
}

__global__ void segexp_kernel(const float* __restrict__ gate, const int* __restrict__ batch,
                              const int* __restrict__ mkey, float* __restrict__ e,
                              float* __restrict__ s) {
  int n = blockIdx.x * blockDim.x + threadIdx.x;
  if (n >= N_NODES) return;
  int b = batch[n];
  float ev = __expf(gate[n] - key2f(mkey[b]));
  e[n] = ev;
  atomicAdd(&s[b], ev);
}

__global__ void pool_kernel(const float* __restrict__ h, const int* __restrict__ batch,
                            const float* __restrict__ e, const float* __restrict__ s,
                            float* __restrict__ pooled) {
  int t = blockIdx.x * blockDim.x + threadIdx.x;
  if (t >= N_NODES * 64) return;
  int n = t >> 6, d = t & 63;
  int b = batch[n];
  atomicAdd(&pooled[b * 64 + d], e[n] * __builtin_amdgcn_rcpf(s[b]) * h[t]);
}

__global__ void outproj_kernel(const float* __restrict__ pooled, const float* __restrict__ W,
                               const float* __restrict__ b, float* __restrict__ out) {
  int t = blockIdx.x * blockDim.x + threadIdx.x;
  if (t >= N_GRAPHS * 16) return;
  int g = t >> 4, o = t & 15;
  float acc = b[o];
#pragma unroll
  for (int k = 0; k < 64; ++k) acc += pooled[g * 64 + k] * W[k * 16 + o];
  out[t] = acc;
}

// ---------------------------------------------------------------------------
extern "C" void kernel_launch(void* const* d_in, const int* in_sizes, int n_in,
                              void* d_out, int out_size, void* d_ws, size_t ws_size,
                              hipStream_t stream) {
  (void)in_sizes; (void)n_in; (void)out_size; (void)ws_size;
  // setup_inputs order: x, pos, edge_index, batch, params(leaves in jax sorted-key order):
  //   4:gate1.W 5:gate1.b 6:gate2.W 7:gate2.b 8:gate3.W 9:gate3.b 10:inW 11:inb
  //   12+15*l: coord1.W,coord1.b,coord2.W,coord2.b,coors_scale,edge1.W,edge1.b,
  //            edge2.W,edge2.b,ewc.W,ewc.b,node1.W,node1.b,node2.W,node2.b
  //   57:ln_beta 58:ln_gamma 59:outW 60:outb
  const float* x   = (const float*)d_in[0];
  const float* pos = (const float*)d_in[1];
  const int*   eix = (const int*)d_in[2];
  const int*   bat = (const int*)d_in[3];
  const int* row = eix;
  const int* col = eix + N_EDGES;
  auto f = [&](int i) { return (const float*)d_in[i]; };

  char* base = (char*)d_ws;
  size_t off = 0;
  auto alloc = [&](size_t bytes) -> void* {
    void* p = base + off;
    off = (off + bytes + 255) & ~(size_t)255;
    return p;
  };
  float* h      = (float*)alloc((size_t)N_NODES * 64 * 4);
  float* coords = (float*)alloc((size_t)N_NODES * 3 * 4);
  float* cdelta = (float*)alloc((size_t)N_NODES * 3 * 4);
  float* agg    = (float*)alloc((size_t)N_NODES * 64 * 4);
  float* gbuf   = (float*)alloc((size_t)N_NODES * 4);
  float* ebuf   = (float*)alloc((size_t)N_NODES * 4);
  int*   mkey   = (int*)alloc(N_GRAPHS * 4);
  float* sbuf   = (float*)alloc(N_GRAPHS * 4);
  float* pooled = (float*)alloc(N_GRAPHS * 64 * 4);
  _Float16* packed = (_Float16*)alloc((size_t)3 * 148 * 512 * 2);

  auto packW = [&](const float* W, _Float16* dst, int Kr, int Nr, int kc, int nt) {
    int thr = kc * nt * 32;
    pack_w_kernel<<<(thr + 255) / 256, 256, 0, stream>>>(W, dst, Kr, Nr, kc, nt);
  };
  // per-layer packed layout (fragment offsets in units of 512 halfs):
  // coord1 @0 (40), coord2 @40 (4), edge1 @44 (40), edge2 @84 (16),
  // node1 @100 (32), node2 @132 (16); 148 frags/layer.
  for (int l = 0; l < 3; ++l) {
    int pb = 12 + l * 15;
    _Float16* lp = packed + (size_t)l * 148 * 512;
    packW(f(pb + 0),  lp + (size_t)0   * 512, 131, 128, 5, 8);  // coord1.W
    packW(f(pb + 2),  lp + (size_t)40  * 512, 128, 3,   4, 1);  // coord2.W
    packW(f(pb + 5),  lp + (size_t)44  * 512, 131, 128, 5, 8);  // edge1.W
    packW(f(pb + 7),  lp + (size_t)84  * 512, 128, 64,  4, 4);  // edge2.W
    packW(f(pb + 11), lp + (size_t)100 * 512, 128, 128, 4, 8);  // node1.W
    packW(f(pb + 13), lp + (size_t)132 * 512, 128, 64,  4, 4);  // node2.W
  }

  copy_kernel<<<(N_NODES * 3 + 255) / 256, 256, 0, stream>>>(coords, pos, N_NODES * 3);
  input_proj_kernel<<<(N_NODES * 64) / 256, 256, 0, stream>>>(x, f(10), f(11), h);

  const int EDGE_BLOCKS = N_EDGES / 128;          // 6250, exact
  const int NODE_BLOCKS = (N_NODES / 16 + 7) / 8; // 391
  for (int l = 0; l < 3; ++l) {
    int pb = 12 + l * 15;
    _Float16* lp = packed + (size_t)l * 148 * 512;
    zero_kernel<<<(N_NODES * 3 + 255) / 256, 256, 0, stream>>>(cdelta, N_NODES * 3);
    coord_kernel<<<EDGE_BLOCKS, 256, 0, stream>>>(
        h, coords, row, col,
        lp + (size_t)0 * 512,  f(pb + 1),
        lp + (size_t)40 * 512, f(pb + 3),
        f(pb + 9), f(pb + 10), f(pb + 4), cdelta);
    addinto_kernel<<<(N_NODES * 3 + 255) / 256, 256, 0, stream>>>(coords, cdelta, N_NODES * 3);
    zero_kernel<<<(N_NODES * 64) / 256, 256, 0, stream>>>(agg, N_NODES * 64);
    edge_msg_kernel<<<EDGE_BLOCKS, 256, 0, stream>>>(
        h, coords, row, col,
        lp + (size_t)44 * 512, f(pb + 6),
        lp + (size_t)84 * 512, f(pb + 8), agg);
    node_kernel<<<NODE_BLOCKS, 256, 0, stream>>>(
        h, agg, lp + (size_t)100 * 512, f(pb + 12), lp + (size_t)132 * 512, f(pb + 14));
  }

  gate_kernel<<<(N_NODES + 7) / 8, 256, 0, stream>>>(
      h, f(4), f(5), f(58), f(57), f(6), f(7), f(8), f(9), gbuf);
  initpool_kernel<<<(N_GRAPHS * 64 + 255) / 256, 256, 0, stream>>>(mkey, sbuf, pooled);
  segmax_kernel<<<(N_NODES + 255) / 256, 256, 0, stream>>>(gbuf, bat, mkey);
  segexp_kernel<<<(N_NODES + 255) / 256, 256, 0, stream>>>(gbuf, bat, mkey, ebuf, sbuf);
  pool_kernel<<<(N_NODES * 64) / 256, 256, 0, stream>>>(h, bat, ebuf, sbuf, pooled);
  outproj_kernel<<<(N_GRAPHS * 16 + 255) / 256, 256, 0, stream>>>(pooled, f(59), f(60), (float*)d_out);
}
